// MoEALU_62380105007222
// MI455X (gfx1250) — matrix operand order repaired and checked by reading, not compiled, not run on hardware
//
#include <hip/hip_runtime.h>

typedef __attribute__((ext_vector_type(16))) _Float16 v16h;
typedef __attribute__((ext_vector_type(8)))  float    v8f;
typedef __attribute__((ext_vector_type(4)))  float    f32x4;

#define TS 144.269504088896340736f  /* 100 * log2(e)  (softmax temp 100) */
#define WAVES_PER_BLOCK 4

#if __has_builtin(__builtin_amdgcn_exp2f)
#define EX2(x) __builtin_amdgcn_exp2f(x)
#else
#define EX2(x) exp2f(x)
#endif

__device__ __forceinline__ void softmax16(float v[16]) {
  float m = v[0];
#pragma unroll
  for (int i = 1; i < 16; ++i) m = fmaxf(m, v[i]);
  float s = 0.f;
#pragma unroll
  for (int i = 0; i < 16; ++i) { float e = EX2((v[i] - m) * TS); v[i] = e; s += e; }
  float inv = 1.0f / s;
#pragma unroll
  for (int i = 0; i < 16; ++i) v[i] *= inv;
}

// h[16], l[16] = high/low nibble-bin sums of one 256-float row (== row @ b2n),
// computed as split-f16 WMMA over a 16-item tile (one wave).
// A k order used here (contiguous, not the ISA table order): lane L<16 holds row M=L, K={0..15};
// lane L+16 holds row M=L, K={16..31}.  B is the 0/1 bin-indicator:
// low bins  -> per-lane one-hot (chunk-invariant); high bins -> per-lane splat.
__device__ __forceinline__ void nibble_sums_wmma(
    const float* __restrict__ row, float* __restrict__ ldsTile,
    int laneHi, int nid, const v16h& Bl, const v16h& Vone, const v16h& Vzero,
    float h[16], float l[16])
{
  v8f accH = {0.f,0.f,0.f,0.f,0.f,0.f,0.f,0.f};
  v8f accL = accH;
#pragma unroll
  for (int c = 0; c < 256; c += 32) {
    const int off = c + laneHi * 16;
    f32x4 g0 = *(const f32x4*)(row + off);
    f32x4 g1 = *(const f32x4*)(row + off + 4);
    f32x4 g2 = *(const f32x4*)(row + off + 8);
    f32x4 g3 = *(const f32x4*)(row + off + 12);
    float vals[16] = {g0.x, g0.y, g0.z, g0.w, g1.x, g1.y, g1.z, g1.w,
                      g2.x, g2.y, g2.z, g2.w, g3.x, g3.y, g3.z, g3.w};
    v16h Ahi, Alo;
#pragma unroll
    for (int e = 0; e < 16; ++e) {
      _Float16 hi = (_Float16)vals[e];
      Ahi[e] = hi;
      Alo[e] = (_Float16)(vals[e] - (float)hi);   // f32 = hi + lo split
    }
    // high-bin indicator: this lane's K-range maps to bin (c>>4)+laneHi
    const v16h Bh = (nid == ((c >> 4) + laneHi)) ? Vone : Vzero;
    accH = __builtin_amdgcn_wmma_f32_16x16x32_f16(false, Alo, false, Bh, (short)0, accH, false, false);
    accH = __builtin_amdgcn_wmma_f32_16x16x32_f16(false, Ahi, false, Bh, (short)0, accH, false, false);
    accL = __builtin_amdgcn_wmma_f32_16x16x32_f16(false, Alo, false, Bl, (short)0, accL, false, false);
    accL = __builtin_amdgcn_wmma_f32_16x16x32_f16(false, Ahi, false, Bl, (short)0, accL, false, false);
  }
  // D layout: elem r of acc, this lane -> (item = r + 8*laneHi, bin = nid)
#pragma unroll
  for (int r = 0; r < 8; ++r) {
    const int m = r + 8 * laneHi;
    ldsTile[m * 32 + nid]      = accH[r];
    ldsTile[m * 32 + 16 + nid] = accL[r];
  }
  __syncthreads();
#pragma unroll
  for (int j = 0; j < 16; j += 4) {
    f32x4 hv = *(const f32x4*)(ldsTile + nid * 32 + j);
    f32x4 lv = *(const f32x4*)(ldsTile + nid * 32 + 16 + j);
    h[j] = hv.x; h[j+1] = hv.y; h[j+2] = hv.z; h[j+3] = hv.w;
    l[j] = lv.x; l[j+1] = lv.y; l[j+2] = lv.z; l[j+3] = lv.w;
  }
  __syncthreads();
}

// Soft ripple nibble add: x,y,cin are probability 16-/2-vectors.
// s_pre = circular conv over Z16; carry mass = sum over i+j+k >= 16.
__device__ __forceinline__ void nibble_add16(
    const float x[16], const float y[16], float& c0, float& c1, float s[16])
{
  float u[31];
#pragma unroll
  for (int t = 0; t < 31; ++t) u[t] = 0.f;
#pragma unroll
  for (int i = 0; i < 16; ++i)
#pragma unroll
    for (int j = 0; j < 16; ++j)
      u[i + j] = fmaf(x[i], y[j], u[i + j]);

  float cr0_k0 = 0.f, cr1_k0 = 0.f, cr0_k1 = 0.f, cr1_k1 = 0.f;
#pragma unroll
  for (int t = 0; t < 31; ++t) {
    if (t < 16) cr0_k0 += u[t]; else cr1_k0 += u[t];
    if (t < 15) cr0_k1 += u[t]; else cr1_k1 += u[t];
  }
#pragma unroll
  for (int d = 0; d < 16; ++d) {
    float s0 = u[d] + ((d + 16 <= 30) ? u[d + 16] : 0.f);          // cin = 0
    float s1 = ((d >= 1) ? u[d - 1] : 0.f) + ((d + 15 <= 30) ? u[d + 15] : 0.f); // cin = 1
    s[d] = c0 * s0 + c1 * s1;
  }
  softmax16(s);
  float ca = c0 * cr0_k0 + c1 * cr0_k1;
  float cb = c0 * cr1_k0 + c1 * cr1_k1;
  float m = fmaxf(ca, cb);
  float e0 = EX2((ca - m) * TS), e1 = EX2((cb - m) * TS);
  float inv = 1.0f / (e0 + e1);
  c0 = e0 * inv; c1 = e1 * inv;
}

__global__ __launch_bounds__(32 * WAVES_PER_BLOCK) void moealu_kernel(
    const float* __restrict__ a, const float* __restrict__ b,
    float* __restrict__ out, int nItems)
{
  __shared__ alignas(16) float lds[WAVES_PER_BLOCK][16 * 32];
  const int tid    = threadIdx.x;
  const int wave   = tid >> 5;
  const int lane   = tid & 31;
  const int laneHi = lane >> 4;     // both half-waves carry the same item
  const int nid    = lane & 15;
  int item = (blockIdx.x * WAVES_PER_BLOCK + wave) * 16 + nid;
  if (item >= nItems) item = nItems - 1;   // benign clamp (exact grid in practice)
  const float* pa = a   + (size_t)item * 1024;
  const float* pb = b   + (size_t)item * 1024;
  float*       po = out + (size_t)item * 1024;

  v16h Vone, Vzero, Bl;
#pragma unroll
  for (int e = 0; e < 16; ++e) {
    Vone[e]  = (_Float16)1.0f;
    Vzero[e] = (_Float16)0.0f;
    Bl[e]    = (e == nid) ? (_Float16)1.0f : (_Float16)0.0f;  // low-bin one-hot
  }

  float c0 = 1.0f, c1 = 0.0f;     // initial carry distribution [1,0]
#pragma unroll 1
  for (int byte = 0; byte < 4; ++byte) {
    float ah[16], al[16], bh[16], bl[16];
    nibble_sums_wmma(pa + byte * 256, lds[wave], laneHi, nid, Bl, Vone, Vzero, ah, al);
    softmax16(ah); softmax16(al);
    nibble_sums_wmma(pb + byte * 256, lds[wave], laneHi, nid, Bl, Vone, Vzero, bh, bl);
    softmax16(bh); softmax16(bl);

    float sl[16], sh[16];
    nibble_add16(al, bl, c0, c1, sl);   // low nibble first (matches reference)
    nibble_add16(ah, bh, c0, c1, sh);

    // from_nibbles factorizes: softmax256(sh[h]+sl[l]) = softmax16(sh) x softmax16(sl)
    softmax16(sh); softmax16(sl);
    float* pout = po + byte * 256;
#pragma unroll
    for (int r = 0; r < 8; ++r) {
      const int hI = 8 * laneHi + r;   // halves split the 16 output rows
      const float sH = sh[hI];
#pragma unroll
      for (int q = 0; q < 4; ++q) {
        f32x4 v = { sH * sl[4*q], sH * sl[4*q+1], sH * sl[4*q+2], sH * sl[4*q+3] };
        *(f32x4*)(pout + hI * 16 + 4 * q) = v;
      }
    }
  }
}

extern "C" void kernel_launch(void* const* d_in, const int* in_sizes, int n_in,
                              void* d_out, int out_size, void* d_ws, size_t ws_size,
                              hipStream_t stream) {
  const float* a = (const float*)d_in[0];
  const float* b = (const float*)d_in[1];
  float* out = (float*)d_out;
  const int nItems = in_sizes[0] / 1024;              // a is [B,4,256]
  const int tiles  = (nItems + 15) / 16;              // 16 items per wave
  const int blocks = (tiles + WAVES_PER_BLOCK - 1) / WAVES_PER_BLOCK;
  moealu_kernel<<<blocks, 32 * WAVES_PER_BLOCK, 0, stream>>>(a, b, out, nItems);
}
